// AttentionPooling_33071248180148
// MI455X (gfx1250) — matrix operand order repaired and checked by reading, not compiled, not run on hardware
//
#include <hip/hip_runtime.h>
#include <hip/hip_bf16.h>

typedef _Float16 h16;
typedef __attribute__((ext_vector_type(16))) _Float16 v16h;
typedef __attribute__((ext_vector_type(8)))  _Float16 v8h;
typedef __attribute__((ext_vector_type(8)))  float    v8f;
typedef unsigned int u32;
typedef __attribute__((ext_vector_type(4))) u32 u32x4;
typedef __attribute__((ext_vector_type(8))) u32 u32x8;

#define BATCH  128
#define DMODEL 512
#define NSEQ   512
#define NQ     16
#define TILE   32
#define NTILES (NSEQ / TILE)
#define SLAT_BYTES (DMODEL * TILE * 4)   // 64 KB per staging buffer

// sqrt(512), 1/sqrt(512), -ln(10000)/512
#define SCALE_X   22.627416997969522f
#define ISQRT_DH  0.044194173824159216f
#define PE_COEF  -0.017988946039015980f

union V16U { v16h v; v8h h[2]; };

__device__ __forceinline__ float grp16_max(float v) {
  v = fmaxf(v, __shfl_xor(v, 1, 32));
  v = fmaxf(v, __shfl_xor(v, 2, 32));
  v = fmaxf(v, __shfl_xor(v, 4, 32));
  v = fmaxf(v, __shfl_xor(v, 8, 32));
  return v;
}
__device__ __forceinline__ float grp16_sum(float v) {
  v += __shfl_xor(v, 1, 32);
  v += __shfl_xor(v, 2, 32);
  v += __shfl_xor(v, 4, 32);
  v += __shfl_xor(v, 8, 32);
  return v;
}

// Issue a TDM load: 2D tile (tile_dim0=32 elems of 4B, tile_dim1=512 rows,
// row stride 512 elems) from global 'gaddr' into LDS byte offset 'lds_off'.
__device__ __forceinline__ void tdm_issue(const float* gaddr, u32 lds_off) {
  unsigned long long ga = (unsigned long long)(size_t)gaddr;
  u32x4 g0;
  g0[0] = 1u;                                    // count=1 (valid descriptor)
  g0[1] = lds_off;                               // lds_addr
  g0[2] = (u32)ga;                               // global_addr[31:0]
  g0[3] = (u32)(ga >> 32) | (2u << 30);          // global_addr[56:32] | type=2
  u32x8 g1;
  g1[0] = 0x20000u;                              // data_size=2 (4B), no mask/pad/iterate
  g1[1] = (512u << 16);                          // tensor_dim0[15:0]=512 in [31:16]
  g1[2] = (512u << 16);                          // tensor_dim0 hi=0 | tensor_dim1[15:0]=512
  g1[3] = (32u << 16);                           // tensor_dim1 hi=0 | tile_dim0=32
  g1[4] = 512u;                                  // tile_dim1=512, tile_dim2=0
  g1[5] = 512u;                                  // tensor_dim0_stride[31:0]=512
  g1[6] = 0u;                                    // stride hi | tensor_dim1_stride lo
  g1[7] = 0u;
  asm volatile("tensor_load_to_lds %0, %1" :: "s"(g0), "s"(g1) : "memory");
}

// ---------------- prep kernels ----------------

__global__ void qln_kernel(const float* __restrict__ query,
                           const float* __restrict__ w, const float* __restrict__ b,
                           float* __restrict__ qln) {
  __shared__ float r1[512];
  __shared__ float r2[512];
  const int i = blockIdx.x, d = threadIdx.x;
  float v = query[i * 512 + d] * SCALE_X;
  r1[d] = v; r2[d] = v * v;
  __syncthreads();
  for (int s = 256; s > 0; s >>= 1) {
    if (d < s) { r1[d] += r1[d + s]; r2[d] += r2[d + s]; }
    __syncthreads();
  }
  float mean = r1[0] * (1.0f / 512.0f);
  float var  = r2[0] * (1.0f / 512.0f) - mean * mean;
  float rstd = rsqrtf(var + 1e-5f);
  qln[i * 512 + d] = (v - mean) * rstd * w[d] + b[d];
}

__global__ void qh_kernel(const float* __restrict__ qln, const float* __restrict__ Wq,
                          const float* __restrict__ bq, h16* __restrict__ qh16) {
  const int i = blockIdx.x, j = threadIdx.x;
  float acc = bq[j];
  for (int d = 0; d < 512; ++d) acc += qln[i * 512 + d] * Wq[j * 512 + d];
  qh16[i * 512 + j] = (h16)acc;
}

__global__ void wt_kernel(const float* __restrict__ Wk, const float* __restrict__ Wv,
                          h16* __restrict__ WkT, h16* __restrict__ WvT) {
  int idx = blockIdx.x * 256 + threadIdx.x;
  int sel = idx >> 18;
  int local = idx & (512 * 512 - 1);
  int d = local >> 9, j = local & 511;
  const float* src = sel ? Wv : Wk;
  h16* dst = sel ? WvT : WkT;
  dst[local] = (h16)src[local];
}

// ---------------- fused attention-pooling kernel ----------------

__global__ __launch_bounds__(256) void attn_main(
    const float* __restrict__ latents, const float* __restrict__ qmask,
    const h16* __restrict__ qh16, const h16* __restrict__ WkT, const h16* __restrict__ WvT,
    const float* __restrict__ bk, const float* __restrict__ bv,
    const float* __restrict__ lnw, const float* __restrict__ lnb,
    const float* __restrict__ Wo, const float* __restrict__ bo,
    float* __restrict__ out_logits, float* __restrict__ out_attnw) {
  extern __shared__ char smem[];
  // TDM staging buffers (double buffered), [512 d][32 n] f32 each
  float* slat0 = (float*)smem;                      // 64 KB
  float* slat1 = (float*)(smem + SLAT_BYTES);       // 64 KB
  // f32 region
  float* ss    = (float*)(smem + 2 * SLAT_BYTES);   // [16][512] raw scores; reused as attn
  float* spart = ss + 16 * 512;                     // [16][16][16] score partials
  float* sred  = spart + 16 * 256;                  // [32][8]
  float* sred2 = sred + 256;                        // [32][8]
  float* smean = sred2 + 256;                       // [32]
  float* srstd = smean + 32;                        // [32]
  float* salpha = srstd + 32;                       // [16]
  float* sm_   = salpha + 16;                       // [16]
  float* sl_   = sm_ + 16;                          // [16]
  float* sqm   = sl_ + 16;                          // [16]
  float* sfreq = sqm + 16;                          // [512]
  float* slnw  = sfreq + 512;                       // [512]
  float* slnb  = slnw + 512;                        // [512]
  // f16 region
  h16* sxh  = (h16*)(slnb + 512);                   // [32][512] normalized x tile
  h16* skhT = sxh + 32 * 512;                       // [32][512] K tile, [seq][d]
  h16* svh  = skhT + 512 * 32;                      // [512][32] V tile transposed, [dh][seq]
  h16* sp   = svh + 32 * 512;                       // [16][32] P tile

  const int b    = blockIdx.x;
  const int tid  = threadIdx.x;
  const int wave = tid >> 5;
  const int lane = tid & 31;
  const int lm   = lane & 15;
  const int lh   = lane >> 4;
  const int tn   = tid & 31;   // seq-in-tile for LN phase
  const int td   = tid >> 5;   // d-chunk (64 wide) for LN phase

  const float* lat = latents + (size_t)b * DMODEL * NSEQ;
  const u32 lds_dyn = __builtin_amdgcn_groupstaticsize();  // dynamic LDS base offset

  // kick off DMA of tile 0 into buffer 0
  if (wave == 0) tdm_issue(lat, lds_dyn);

  // batch-start init
  for (int d = tid; d < 512; d += 256) {
    sfreq[d] = __expf((float)(d & ~1) * PE_COEF);
    slnw[d] = lnw[d];
    slnb[d] = lnb[d];
  }
  if (tid < NQ) {
    sm_[tid] = -1e30f;
    sl_[tid] = 0.0f;
    sqm[tid] = qmask[b * NQ + tid];
  }

  v8f oacc[4];
  #pragma unroll
  for (int nt = 0; nt < 4; ++nt)
    #pragma unroll
    for (int v = 0; v < 8; ++v) oacc[nt][v] = 0.0f;

  for (int t = 0; t < NTILES; ++t) {
    // ---- TDM pipeline: issue t+1, wait for t ----
    if (wave == 0) {
      if (t + 1 < NTILES) {
        tdm_issue(lat + (size_t)(t + 1) * TILE, lds_dyn + (u32)(((t + 1) & 1) * SLAT_BYTES));
        __builtin_amdgcn_s_wait_tensorcnt(1);   // tile t resident (in-order completion)
      } else {
        __builtin_amdgcn_s_wait_tensorcnt(0);
      }
    }
    __syncthreads();
    const float* slat = (t & 1) ? slat1 : slat0;

    // ---- Phase A: x = LN(latents^T * scale + reversed-PE) -> sxh (f16) ----
    {
      const int nglob = t * TILE + tn;
      const float p = (float)(NSEQ - 1 - nglob);
      float sum = 0.0f, sum2 = 0.0f;
      for (int it = 0; it < 64; ++it) {
        int d = td * 64 + it;
        float ang = p * sfreq[d];
        float pe = (d & 1) ? __cosf(ang) : __sinf(ang);
        float v = slat[d * TILE + tn] * SCALE_X + pe;
        sxh[tn * 512 + d] = (h16)v;
        sum += v; sum2 += v * v;
      }
      sred[tn * 8 + td] = sum;
      sred2[tn * 8 + td] = sum2;
    }
    __syncthreads();
    if (tid < 32) {
      float sum = 0.0f, sum2 = 0.0f;
      for (int k = 0; k < 8; ++k) { sum += sred[tid * 8 + k]; sum2 += sred2[tid * 8 + k]; }
      float mean = sum * (1.0f / 512.0f);
      float var  = sum2 * (1.0f / 512.0f) - mean * mean;
      smean[tid] = mean;
      srstd[tid] = rsqrtf(var + 1e-5f);
    }
    __syncthreads();
    {
      float mean = smean[tn], rstd = srstd[tn];
      for (int it = 0; it < 64; ++it) {
        int d = td * 64 + it;
        float v = (float)sxh[tn * 512 + d];
        sxh[tn * 512 + d] = (h16)((v - mean) * rstd * slnw[d] + slnb[d]);
      }
    }
    __syncthreads();

    // ---- Phase B: K/V projection (wave owns a 64-wide dh slice; A reused 8x) ----
    const int dhw = wave * 64;
    for (int mh = 0; mh < 2; ++mh) {
      v8f kacc[4], vacc[4];
      #pragma unroll
      for (int nt = 0; nt < 4; ++nt)
        #pragma unroll
        for (int v = 0; v < 8; ++v) { kacc[nt][v] = 0.0f; vacc[nt][v] = 0.0f; }
      for (int kt = 0; kt < 16; ++kt) {
        V16U a;
        const h16* arow = sxh + (mh * 16 + lm) * 512 + kt * 32 + lh * 8;
        a.h[0] = *(const v8h*)(arow);
        a.h[1] = *(const v8h*)(arow + 16);
        const h16* wrow_k = WkT + (size_t)(dhw + lm) * 512 + kt * 32 + lh * 8;
        const h16* wrow_v = WvT + (size_t)(dhw + lm) * 512 + kt * 32 + lh * 8;
        #pragma unroll
        for (int nt = 0; nt < 4; ++nt) {
          V16U bk_, bv_;
          bk_.h[0] = *(const v8h*)(wrow_k + nt * 16 * 512);
          bk_.h[1] = *(const v8h*)(wrow_k + nt * 16 * 512 + 16);
          bv_.h[0] = *(const v8h*)(wrow_v + nt * 16 * 512);
          bv_.h[1] = *(const v8h*)(wrow_v + nt * 16 * 512 + 16);
          kacc[nt] = __builtin_amdgcn_wmma_f32_16x16x32_f16(false, a.v, false, bk_.v, (short)0, kacc[nt], false, false);
          vacc[nt] = __builtin_amdgcn_wmma_f32_16x16x32_f16(false, a.v, false, bv_.v, (short)0, vacc[nt], false, false);
        }
      }
      #pragma unroll
      for (int nt = 0; nt < 4; ++nt) {
        const int col = dhw + nt * 16 + lm;
        const float bkc = bk[col], bvc = bv[col];
        v8h vst;
        #pragma unroll
        for (int v = 0; v < 8; ++v) vst[v] = (h16)(vacc[nt][v] + bvc);
        *(v8h*)(svh + col * 32 + mh * 16 + lh * 8) = vst;   // transposed: [dh][seq]
        #pragma unroll
        for (int v = 0; v < 8; ++v) {
          const int row = mh * 16 + v + 8 * lh;
          skhT[row * 512 + col] = (h16)(kacc[nt][v] + bkc);   // [seq][d]
        }
      }
    }
    __syncthreads();

    // ---- Phase C: score partials, distributed over all 8 waves ----
    {
      v8f sacc[2];
      #pragma unroll
      for (int nh = 0; nh < 2; ++nh)
        #pragma unroll
        for (int v = 0; v < 8; ++v) sacc[nh][v] = 0.0f;
      for (int kk = 0; kk < 2; ++kk) {
        const int kt = wave * 2 + kk;
        V16U a;
        const h16* qrow = qh16 + lm * 512 + kt * 32 + lh * 8;
        a.h[0] = *(const v8h*)(qrow);
        a.h[1] = *(const v8h*)(qrow + 16);
        const h16* krow = skhT + lm * 512 + kt * 32 + lh * 8;
        #pragma unroll
        for (int nh = 0; nh < 2; ++nh) {
          V16U bf;
          bf.h[0] = *(const v8h*)(krow + nh * 16 * 512);
          bf.h[1] = *(const v8h*)(krow + nh * 16 * 512 + 16);
          sacc[nh] = __builtin_amdgcn_wmma_f32_16x16x32_f16(false, a.v, false, bf.v, (short)0, sacc[nh], false, false);
        }
      }
      #pragma unroll
      for (int nh = 0; nh < 2; ++nh)
        #pragma unroll
        for (int v = 0; v < 8; ++v)
          spart[(wave * 2 + nh) * 256 + (v + 8 * lh) * 16 + lm] = sacc[nh][v];
    }
    __syncthreads();

    // ---- Phase D: reduce + mask + online softmax (wave 0) ----
    if (wave == 0) {
      float sv[2][8];
      #pragma unroll
      for (int nh = 0; nh < 2; ++nh) {
        #pragma unroll
        for (int v = 0; v < 8; ++v) {
          const int row = v + 8 * lh;
          float s = 0.0f;
          for (int w = 0; w < 8; ++w) s += spart[(w * 2 + nh) * 256 + row * 16 + lm];
          s *= ISQRT_DH;
          if (sqm[row] == 0.0f) s = -1e9f;
          sv[nh][v] = s;
          ss[row * 512 + t * TILE + nh * 16 + lm] = s;
        }
      }
      #pragma unroll
      for (int v = 0; v < 8; ++v) {
        const int row = v + 8 * lh;
        const float mrow = fmaxf(grp16_max(sv[0][v]), grp16_max(sv[1][v]));
        const float mo = sm_[row];
        const float mn = fmaxf(mo, mrow);
        const float p0 = __expf(sv[0][v] - mn);
        const float p1 = __expf(sv[1][v] - mn);
        const float rs = grp16_sum(p0 + p1);
        if (lm == 0) {
          const float al = __expf(mo - mn);
          salpha[row] = al;
          sl_[row] = sl_[row] * al + rs;
          sm_[row] = mn;
        }
        sp[row * 32 + lm] = (h16)p0;
        sp[row * 32 + 16 + lm] = (h16)p1;
      }
    }
    __syncthreads();

    // ---- Phase E: rescale accumulators, P @ V (full K=32) ----
    {
      V16U a;
      const h16* prow = sp + lm * 32 + lh * 8;
      a.h[0] = *(const v8h*)(prow);
      a.h[1] = *(const v8h*)(prow + 16);
      float al[8];
      #pragma unroll
      for (int v = 0; v < 8; ++v) al[v] = salpha[v + 8 * lh];
      const h16* vrow = svh + (dhw + lm) * 32 + lh * 8;
      #pragma unroll
      for (int nt = 0; nt < 4; ++nt) {
        V16U bf;
        bf.h[0] = *(const v8h*)(vrow + nt * 16 * 32);
        bf.h[1] = *(const v8h*)(vrow + nt * 16 * 32 + 16);
        #pragma unroll
        for (int v = 0; v < 8; ++v) oacc[nt][v] *= al[v];
        oacc[nt] = __builtin_amdgcn_wmma_f32_16x16x32_f16(false, a.v, false, bf.v, (short)0, oacc[nt], false, false);
      }
    }
  }
  __syncthreads();

  // attn_weights output: exp(s - m_final) / l_final
  {
    float* oat = out_attnw + (size_t)b * NQ * NSEQ;
    for (int i = tid; i < NQ * NSEQ; i += 256) {
      const int q = i >> 9;
      oat[i] = __expf(ss[i] - sm_[q]) / sl_[q];
    }
  }
  __syncthreads();
  // attn values (normalized + wiped) into ss (reuse)
  for (int nt = 0; nt < 4; ++nt) {
    const int col = wave * 64 + nt * 16 + lm;
    #pragma unroll
    for (int v = 0; v < 8; ++v) {
      const int row = v + 8 * lh;
      const float val = (sqm[row] == 0.0f) ? 0.0f : oacc[nt][v] / sl_[row];
      ss[row * 512 + col] = val;
    }
  }
  __syncthreads();

  // logits = attn @ Wo^T + bo : wave w handles query rows 2w, 2w+1
  for (int r = wave * 2; r < wave * 2 + 2; ++r) {
    float part = 0.0f;
    for (int d = lane; d < 512; d += 32) part += ss[r * 512 + d] * Wo[d];
    part += __shfl_xor(part, 16, 32);
    part += __shfl_xor(part, 8, 32);
    part += __shfl_xor(part, 4, 32);
    part += __shfl_xor(part, 2, 32);
    part += __shfl_xor(part, 1, 32);
    if (lane == 0) out_logits[b * NQ + r] = part + bo[0];
  }
}

extern "C" void kernel_launch(void* const* d_in, const int* in_sizes, int n_in,
                              void* d_out, int out_size, void* d_ws, size_t ws_size,
                              hipStream_t stream) {
  const float* latents  = (const float*)d_in[0];
  const float* qmask    = (const float*)d_in[1];
  const float* query    = (const float*)d_in[2];
  const float* ln_lat_w = (const float*)d_in[3];
  const float* ln_lat_b = (const float*)d_in[4];
  const float* ln_q_w   = (const float*)d_in[5];
  const float* ln_q_b   = (const float*)d_in[6];
  const float* Wq       = (const float*)d_in[7];
  const float* bq       = (const float*)d_in[8];
  const float* Wk       = (const float*)d_in[9];
  const float* bk       = (const float*)d_in[10];
  const float* Wv       = (const float*)d_in[11];
  const float* bv       = (const float*)d_in[12];
  const float* Wo       = (const float*)d_in[13];
  const float* bo       = (const float*)d_in[14];

  float* out_logits = (float*)d_out;                 // [B, 16, 1]
  float* out_attnw  = out_logits + BATCH * NQ;       // [B, 1, 16, 512]

  // workspace: qh16 (16K) | WkT (512K) | WvT (512K) | qln (32K)
  h16* qh16 = (h16*)d_ws;
  h16* WkT  = qh16 + NQ * DMODEL;
  h16* WvT  = WkT + DMODEL * DMODEL;
  float* qln = (float*)(WvT + DMODEL * DMODEL);

  qln_kernel<<<NQ, 512, 0, stream>>>(query, ln_q_w, ln_q_b, qln);
  qh_kernel<<<NQ, 512, 0, stream>>>(qln, Wq, bq, qh16);
  wt_kernel<<<2048, 256, 0, stream>>>(Wk, Wv, WkT, WvT);

  // LDS: 2x64KB TDM staging + 14464 f32 + 49664 f16 = 288,256 B (< 320 KB/WGP)
  const size_t smem = (size_t)2 * SLAT_BYTES
                    + (size_t)14464 * sizeof(float)
                    + (size_t)49664 * sizeof(h16);
  attn_main<<<BATCH, 256, smem, stream>>>(latents, qmask, qh16, WkT, WvT,
                                          bk, bv, ln_lat_w, ln_lat_b, Wo, bo,
                                          out_logits, out_attnw);
}
